// Quantization_41446434406895
// MI455X (gfx1250) — matrix operand order repaired and checked by reading, not compiled, not run on hardware
//
#include <hip/hip_runtime.h>

// ---------------------------------------------------------------------------
// VQ nearest-codebook kernel for MI455X (gfx1250, wave32, WMMA).
//   dist ranking:  ||c||^2 - 2*x.c   (||x||^2 is row-constant -> dropped)
//   x.c computed with split-bf16 WMMA: x_hi*c_hi + x_hi*c_lo + x_lo*c_hi
//   exact loss / emb_out recomputed in a light epilogue pass.
// Wave grid: 8 waves = 2(M) x 4(N); each wave owns two 16x16 accumulators
// (32 rows x 16 cols) so each staged/converted B fragment feeds 6 WMMAs.
// ---------------------------------------------------------------------------

typedef __bf16 v16bf __attribute__((ext_vector_type(16)));
typedef __bf16 v8bf  __attribute__((ext_vector_type(8)));
typedef float  v8f   __attribute__((ext_vector_type(8)));

constexpr int N_ROWS  = 32768;
constexpr int DIM     = 512;
constexpr int K_CODES = 8192;

constexpr int BM = 64;               // rows per block
constexpr int BN = 64;               // codebook columns per tile (4 waves x 16)
constexpr int BK = 32;               // K-depth per WMMA chunk (bf16 16x16x32)
constexpr int TILES_N = K_CODES / BN;   // 128
constexpr int CHUNKS  = DIM / BK;       // 16

// dynamic LDS layout (bytes):
//   xhi  : BM*DIM bf16      = 65536
//   xlo  : BM*DIM bf16      = 65536
//   bstg : 2 * BN*BK f32    = 16384   (double-buffered codebook chunk)
//   red  : 4 * BM * (f32+i32) = 2048
constexpr size_t SMEM_BYTES =
    (size_t)2 * BM * DIM * 2 + (size_t)2 * BN * BK * 4 + (size_t)4 * BM * 8;

// -------- async global->LDS path (guarded; falls back to plain loads) ------
#if defined(__HIP_DEVICE_COMPILE__) &&                                        \
    __has_builtin(__builtin_amdgcn_global_load_async_to_lds_b128) &&          \
    __has_builtin(__builtin_amdgcn_s_wait_asynccnt)
#define USE_ASYNC 1
// builtin signature: (v4i addrspace(1)*, v4i addrspace(3)*, Imm off, Imm cpol)
typedef int v4i_t __attribute__((vector_size(16)));
typedef __attribute__((address_space(1))) v4i_t* gv4i_p;
typedef __attribute__((address_space(3))) v4i_t* lv4i_p;
#else
#define USE_ASYNC 0
#endif

// ---------------------------------------------------------------------------
// Kernel 1: per-code squared norms.  One wave per codebook row.
// ---------------------------------------------------------------------------
__global__ __launch_bounds__(256) void vq_cnorm_kernel(
    const float* __restrict__ codebook, float* __restrict__ cnorm) {
  const int wave = threadIdx.x >> 5, lane = threadIdx.x & 31;
  const int row = blockIdx.x * 8 + wave;
  const float4* src = (const float4*)(codebook + (size_t)row * DIM);
  float s = 0.f;
  for (int i = lane; i < DIM / 4; i += 32) {
    float4 v = src[i];
    s += v.x * v.x + v.y * v.y + v.z * v.z + v.w * v.w;
  }
#pragma unroll
  for (int off = 16; off > 0; off >>= 1) s += __shfl_xor(s, off, 32);
  if (lane == 0) cnorm[row] = s;
}

// ---------------------------------------------------------------------------
// Kernel 2: split-bf16 WMMA distance + argmin.  Grid = N/BM blocks, 8 waves.
// ---------------------------------------------------------------------------
__global__ __launch_bounds__(256) void vq_argmin_kernel(
    const float* __restrict__ x, const float* __restrict__ codebook,
    const float* __restrict__ cnorm, int* __restrict__ ids) {
  extern __shared__ __align__(16) char smem_raw[];
  __bf16* xhi   = (__bf16*)smem_raw;                 // [BM][DIM]
  __bf16* xlo   = xhi + BM * DIM;                    // [BM][DIM]
  float*  bstg  = (float*)(xlo + BM * DIM);          // [2][BN][BK]
  float*  redV  = bstg + 2 * BN * BK;                // [4][BM]
  int*    redI  = (int*)(redV + 4 * BM);             // [4][BM]

  const int tid  = threadIdx.x;
  const int lane = tid & 31;
  const int wave = tid >> 5;
  const int half = lane >> 4;   // 0/1 : which 16-lane half
  const int l16  = lane & 15;
  const int wm   = wave & 1;    // 0..1 : 32-row slice
  const int wn   = wave >> 1;   // 0..3 : 16-col slice
  const int rowBase = blockIdx.x * BM;

  // ---- stage x tile to LDS as bf16 hi/lo (read once per block) ----
  {
    const float4* src = (const float4*)(x + (size_t)rowBase * DIM);
#pragma unroll 4
    for (int i = tid; i < BM * DIM / 4; i += 256) {
      float4 v = src[i];
      const int o = i * 4;
      float f[4] = {v.x, v.y, v.z, v.w};
#pragma unroll
      for (int j = 0; j < 4; ++j) {
        __bf16 h = (__bf16)f[j];
        xhi[o + j] = h;
        xlo[o + j] = (__bf16)(f[j] - (float)h);
      }
    }
  }
  __syncthreads();

  // ---- codebook chunk stager: BN x BK f32 tile, two b128 per thread ----
  auto stageB = [&](int buf, int cb, int kc) {
#pragma unroll
    for (int s = 0; s < 2; ++s) {
      const int idx = tid + s * 256;     // 0..511 float4 slots
      const int c = idx >> 3;            // 0..63 column within tile
      const int k = (idx & 7) * 4;       // 0..28 k within chunk
      const float* gsrc = codebook + (size_t)(cb + c) * DIM + kc + k;
      float* dst = bstg + buf * (BN * BK) + c * BK + k;
#if USE_ASYNC
      __builtin_amdgcn_global_load_async_to_lds_b128(
          (gv4i_p)(void*)gsrc, (lv4i_p)(void*)dst, 0, 0);
#else
      *(float4*)dst = *(const float4*)gsrc;
#endif
    }
  };

  float bestV[2][8];
  int   bestI[2][8];
#pragma unroll
  for (int a = 0; a < 2; ++a)
#pragma unroll
    for (int r = 0; r < 8; ++r) { bestV[a][r] = 3.4e38f; bestI[a][r] = 0; }

  union AFrag { v16bf v; v8bf h[2]; };

#pragma unroll 1
  for (int t = 0; t < TILES_N; ++t) {
    const int cb = t * BN;
    if (t + 1 < TILES_N)  // warm L0/L2 for the next tile
      __builtin_prefetch(codebook + (size_t)(cb + BN) * DIM + tid * 16, 0, 1);

    v8f acc[2];
#pragma unroll
    for (int a = 0; a < 2; ++a)
      acc[a] = (v8f){0.f, 0.f, 0.f, 0.f, 0.f, 0.f, 0.f, 0.f};

    // consume one staged BN x BK chunk: 1 B-fragment conversion -> 6 WMMAs
    auto compute = [&](int buf, int kc) {
      // B fragment (32x16 bf16): lane col n = wn*16 + l16,
      // elements 0..15 <- k = half*16 + 0..15 (contiguous f32 in stage)
      const float* pb =
          bstg + buf * (BN * BK) + (wn * 16 + l16) * BK + half * 16;
      const float4* pb4 = (const float4*)pb;
      float4 b4[4] = {pb4[0], pb4[1], pb4[2], pb4[3]};
      const float* bfp = (const float*)b4;
      v16bf bhi, blo;
#pragma unroll
      for (int e = 0; e < 16; ++e) {
        float f = bfp[e];
        __bf16 h = (__bf16)f;
        bhi[e] = h;
        blo[e] = (__bf16)(f - (float)h);
      }
#pragma unroll
      for (int a = 0; a < 2; ++a) {
        // A fragment (16x32 bf16): lane row m, elements 0..15 <- k=half*16+0..15
        // (two contiguous 16B LDS reads)
        const int m = wm * 32 + a * 16 + l16;
        const __bf16* pah = xhi + m * DIM + kc + half * 16;
        const __bf16* pal = xlo + m * DIM + kc + half * 16;
        AFrag ahi, alo;
        ahi.h[0] = *(const v8bf*)pah;
        ahi.h[1] = *(const v8bf*)(pah + 8);
        alo.h[0] = *(const v8bf*)pal;
        alo.h[1] = *(const v8bf*)(pal + 8);
        acc[a] = __builtin_amdgcn_wmma_f32_16x16x32_bf16(
            false, ahi.v, false, bhi, (short)0, acc[a], false, false);
        acc[a] = __builtin_amdgcn_wmma_f32_16x16x32_bf16(
            false, ahi.v, false, blo, (short)0, acc[a], false, false);
        acc[a] = __builtin_amdgcn_wmma_f32_16x16x32_bf16(
            false, alo.v, false, bhi, (short)0, acc[a], false, false);
      }
    };

    // software pipeline: branch-free steady state + peeled last chunk
    stageB(0, cb, 0);
#pragma unroll 2
    for (int kci = 0; kci < CHUNKS - 1; ++kci) {
      const int buf = kci & 1;
      stageB(buf ^ 1, cb, (kci + 1) * BK);
#if USE_ASYNC
      __builtin_amdgcn_s_wait_asynccnt(2);  // 2 ops of next chunk in flight
#endif
      __syncthreads();
      compute(buf, kci * BK);
      __syncthreads();
    }
    {
#if USE_ASYNC
      __builtin_amdgcn_s_wait_asynccnt(0);
#endif
      __syncthreads();
      compute((CHUNKS - 1) & 1, (CHUNKS - 1) * BK);
      __syncthreads();
    }

    // rank by ||c||^2 - 2*x.c ; lane's column is fixed across all rows
    const int n = cb + wn * 16 + l16;
    const float cn = cnorm[n];
#pragma unroll
    for (int a = 0; a < 2; ++a)
#pragma unroll
      for (int r = 0; r < 8; ++r) {
        float d = cn - 2.0f * acc[a][r];
        if (d < bestV[a][r]) { bestV[a][r] = d; bestI[a][r] = n; }
      }
  }

  // ---- argmin reduce across the 16 lanes sharing a row ----
#pragma unroll
  for (int a = 0; a < 2; ++a)
#pragma unroll
    for (int r = 0; r < 8; ++r) {
      float v = bestV[a][r];
      int   i = bestI[a][r];
#pragma unroll
      for (int off = 8; off > 0; off >>= 1) {
        float ov = __shfl_xor(v, off, 32);
        int   oi = __shfl_xor(i, off, 32);
        if (ov < v || (ov == v && oi < i)) { v = ov; i = oi; }
      }
      bestV[a][r] = v;
      bestI[a][r] = i;
    }

  // ---- combine the four wn column-groups through LDS ----
  if (l16 == 0) {
#pragma unroll
    for (int a = 0; a < 2; ++a)
#pragma unroll
      for (int r = 0; r < 8; ++r) {
        const int row = wm * 32 + a * 16 + r + 8 * half;  // accumulator row map
        redV[wn * BM + row] = bestV[a][r];
        redI[wn * BM + row] = bestI[a][r];
      }
  }
  __syncthreads();
  if (tid < BM) {
    float bv = redV[tid];
    int   bi = redI[tid];
#pragma unroll
    for (int g = 1; g < 4; ++g) {
      float v = redV[g * BM + tid];
      int   i = redI[g * BM + tid];
      if (v < bv || (v == bv && i < bi)) { bv = v; bi = i; }
    }
    ids[rowBase + tid] = bi;
  }
}

// ---------------------------------------------------------------------------
// Kernel 3: gather + exact loss.  One wave per row.
//   emb_out = codebook[id]   (STE forward value)
//   loss    = 1.25 * sum((x - emb)^2)
// ---------------------------------------------------------------------------
__global__ __launch_bounds__(256) void vq_epilogue_kernel(
    const float* __restrict__ x, const float* __restrict__ codebook,
    const int* __restrict__ ids_int, float* __restrict__ emb_out,
    float* __restrict__ ids_out, float* __restrict__ loss_out) {
  const int wave = threadIdx.x >> 5, lane = threadIdx.x & 31;
  const int row = blockIdx.x * 8 + wave;
  const int id = ids_int[row];
  const float4* xp = (const float4*)(x + (size_t)row * DIM);
  const float4* cp = (const float4*)(codebook + (size_t)id * DIM);
  float4* op = (float4*)(emb_out + (size_t)row * DIM);
  float s = 0.f;
  for (int i = lane; i < DIM / 4; i += 32) {
    float4 xv = xp[i], cv = cp[i];
    op[i] = cv;
    float dx = xv.x - cv.x, dy = xv.y - cv.y, dz = xv.z - cv.z, dw = xv.w - cv.w;
    s += dx * dx + dy * dy + dz * dz + dw * dw;
  }
#pragma unroll
  for (int off = 16; off > 0; off >>= 1) s += __shfl_xor(s, off, 32);
  if (lane == 0) {
    loss_out[row] = 1.25f * s;
    ids_out[row] = (float)id;
  }
}

// ---------------------------------------------------------------------------
extern "C" void kernel_launch(void* const* d_in, const int* in_sizes, int n_in,
                              void* d_out, int out_size, void* d_ws,
                              size_t ws_size, hipStream_t stream) {
  (void)in_sizes; (void)n_in; (void)out_size; (void)ws_size;
  const float* x        = (const float*)d_in[0];
  const float* codebook = (const float*)d_in[1];

  float* cnorm   = (float*)d_ws;                                  // K floats
  int*   ids_int = (int*)((char*)d_ws + (size_t)K_CODES * 4);     // N ints

  float* emb_out  = (float*)d_out;                                // N*D
  float* ids_out  = emb_out + (size_t)N_ROWS * DIM;               // N
  float* loss_out = ids_out + N_ROWS;                             // N

  vq_cnorm_kernel<<<K_CODES / 8, 256, 0, stream>>>(codebook, cnorm);

  (void)hipFuncSetAttribute(reinterpret_cast<const void*>(vq_argmin_kernel),
                            hipFuncAttributeMaxDynamicSharedMemorySize,
                            (int)SMEM_BYTES);
  vq_argmin_kernel<<<N_ROWS / BM, 256, SMEM_BYTES, stream>>>(x, codebook,
                                                             cnorm, ids_int);

  vq_epilogue_kernel<<<N_ROWS / 8, 256, 0, stream>>>(x, codebook, ids_int,
                                                     emb_out, ids_out,
                                                     loss_out);
}
